// LoraMultiHeadedSelfAttention_11553462026617
// MI455X (gfx1250) — matrix operand order repaired and checked by reading, not compiled, not run on hardware
//
#include <hip/hip_runtime.h>
#include <hip/hip_bf16.h>

// ---------------------------------------------------------------------------
// Problem constants (from reference): B=2, S=2048, D=768, H=12, R=16, W=64
// ---------------------------------------------------------------------------
#define B_  2
#define S_  2048
#define D_  768
#define H_  12
#define R_  16
#define HW_ 64              // head width
#define LORA_SCALE (1.0f / 16.0f)
#define QSCALE     (0.125f) // 1/sqrt(64), folded into Q projection

typedef __attribute__((ext_vector_type(16))) __bf16        v16bf;
typedef __attribute__((ext_vector_type(16))) unsigned short v16u16;
typedef __attribute__((ext_vector_type(8)))  unsigned short u16x8;
typedef __attribute__((ext_vector_type(8)))  float          v8f;
typedef int v4i_ __attribute__((vector_size(16)));          // matches builtin sig

// gfx1250 async global->LDS copies (ASYNCcnt path); fall back to sync if the
// toolchain doesn't declare the builtins.
#if __has_builtin(__builtin_amdgcn_global_load_async_to_lds_b128) && \
    __has_builtin(__builtin_amdgcn_s_wait_asynccnt)
#define USE_ASYNC 1
#else
#define USE_ASYNC 0
#endif

__device__ __forceinline__ void cp16_g2l(const unsigned short* g, unsigned short* l) {
#if USE_ASYNC
  __builtin_amdgcn_global_load_async_to_lds_b128(
      (__attribute__((address_space(1))) v4i_*)(g),
      (__attribute__((address_space(3))) v4i_*)(l), 0, 0);
#else
  *(u16x8*)l = *(const u16x8*)g;
#endif
}
__device__ __forceinline__ void async_fence() {
#if USE_ASYNC
  __builtin_amdgcn_s_wait_asynccnt(0);
#endif
}

__device__ __forceinline__ unsigned short f2bf(float f) {
  unsigned u = __builtin_bit_cast(unsigned, f);
  u += 0x7FFFu + ((u >> 16) & 1u);          // round-to-nearest-even
  return (unsigned short)(u >> 16);
}

struct U16Pair { u16x8 lo, hi; };
__device__ __forceinline__ v16bf make_frag(u16x8 lo, u16x8 hi) {
  U16Pair p{lo, hi};
  return __builtin_bit_cast(v16bf, p);
}

// A-matrix frag (16x32 bf16) from row-major (M x K) source, ld in elements.
// gfx1250 layout: lanes 0-15 -> M=lane, K 0-7 & 16-23; lanes 16-31 -> K 8-15 & 24-31.
__device__ __forceinline__ v16bf load_a_frag(const unsigned short* base, int ld, int lane) {
  int m  = lane & 15;
  int h8 = (lane >> 4) << 3;                // 0 or 8
  const unsigned short* p = base + m * ld;
  u16x8 lo = *(const u16x8*)(p + h8);       // K = h8 .. h8+7
  u16x8 hi = *(const u16x8*)(p + 16 + h8);  // K = 16+h8 ..
  return make_frag(lo, hi);
}

// B-matrix frag (32x16 bf16) where source is row-major (N x K) (i.e. B^T), ld elems.
// gfx1250 layout: lanes 0-15 -> N=lane, K 0-7 & 16-23; lanes 16-31 -> K 8-15 & 24-31.
__device__ __forceinline__ v16bf load_b_frag(const unsigned short* base, int ld, int lane) {
  int n   = lane & 15;
  int h16 = (lane >> 4) << 3;               // 0 or 8
  const unsigned short* p = base + n * ld + h16;
  u16x8 lo = *(const u16x8*)(p);
  u16x8 hi = *(const u16x8*)(p + 16);
  return make_frag(lo, hi);
}

__device__ __forceinline__ v8f wmma_bf16(v16bf a, v16bf b, v8f c) {
  return __builtin_amdgcn_wmma_f32_16x16x32_bf16(false, a, false, b, (short)0, c,
                                                 false, false);
}

// ---------------------------------------------------------------------------
// Kernel 1: fold LoRA into weights, convert to bf16.
//   W_eff = W + (1/R) * Bm @ A   (rank-16 update; only for proj 0 (q) and 2 (v))
// ---------------------------------------------------------------------------
__global__ __launch_bounds__(256)
void prep_weights(const float* __restrict__ Wq, const float* __restrict__ Aq,
                  const float* __restrict__ Bq, const float* __restrict__ Wk,
                  const float* __restrict__ Wv, const float* __restrict__ Av,
                  const float* __restrict__ Bv, unsigned short* __restrict__ Wb) {
  int idx  = blockIdx.x * blockDim.x + threadIdx.x;   // 0 .. D*D-1
  int proj = blockIdx.y;                              // 0=q 1=k 2=v
  int n = idx / D_, k = idx % D_;
  const float* Ws = (proj == 0) ? Wq : (proj == 1) ? Wk : Wv;
  float acc = Ws[idx];
  if (proj != 1) {
    const float* A  = (proj == 0) ? Aq : Av;
    const float* Bm = (proj == 0) ? Bq : Bv;
    float s = 0.f;
#pragma unroll
    for (int r = 0; r < R_; ++r) s += Bm[n * R_ + r] * A[r * D_ + k];
    acc += s * LORA_SCALE;
  }
  Wb[(size_t)proj * D_ * D_ + idx] = f2bf(acc);
}

// ---------------------------------------------------------------------------
// Kernel 2: x (f32) -> bf16
// ---------------------------------------------------------------------------
__global__ __launch_bounds__(256)
void convert_x(const float* __restrict__ x, unsigned short* __restrict__ xb) {
  int i = blockIdx.x * blockDim.x + threadIdx.x;
  xb[i] = f2bf(x[i]);
}

// ---------------------------------------------------------------------------
// Kernel 3: projection GEMM  Y = xb @ Wb^T + bias   (bf16 in, bf16 out, f32 acc)
// grid (D/128, M/128, 3), 256 threads = 8 waves; wave tile 32(M) x 64(N).
// Double-buffered LDS; next tile fetched with async global->LDS copies while
// the current tile's WMMAs run. Q (proj 0) additionally scaled by 1/sqrt(HW).
// ---------------------------------------------------------------------------
#define BM 128
#define BN 128
#define BK 32
__global__ __launch_bounds__(256)
void proj_gemm(const unsigned short* __restrict__ xb,
               const unsigned short* __restrict__ Wb_all,
               const float* __restrict__ bq, const float* __restrict__ bk,
               const float* __restrict__ bv, unsigned short* __restrict__ Y_all) {
  __shared__ unsigned short As[2][BM][BK];   // x tile   (M x K)
  __shared__ unsigned short Bs[2][BN][BK];   // W tile   (N x K)  == B^T layout

  int proj = blockIdx.z;
  const unsigned short* Wb = Wb_all + (size_t)proj * D_ * D_;
  unsigned short* Y        = Y_all  + (size_t)proj * (B_ * S_) * D_;
  const float* bias = (proj == 0) ? bq : (proj == 1) ? bk : bv;

  int m0 = blockIdx.y * BM;
  int n0 = blockIdx.x * BN;
  int tid = threadIdx.x, wave = tid >> 5, lane = tid & 31;
  int wm = wave & 3, wn = wave >> 2;      // 4 x 2 wave grid

  v8f acc[2][4];
#pragma unroll
  for (int t = 0; t < 2; ++t)
#pragma unroll
    for (int u = 0; u < 4; ++u) acc[t][u] = (v8f){0,0,0,0,0,0,0,0};

  int lrow = tid >> 1;                    // 0..127
  int lcol = (tid & 1) * 16;              // 0 or 16
  const unsigned short* sxr = &xb[(size_t)(m0 + lrow) * D_ + lcol];
  const unsigned short* swr = &Wb[(size_t)(n0 + lrow) * D_ + lcol];

  // prologue: fetch tile 0 into buffer 0
  cp16_g2l(sxr,     &As[0][lrow][lcol]);
  cp16_g2l(sxr + 8, &As[0][lrow][lcol + 8]);
  cp16_g2l(swr,     &Bs[0][lrow][lcol]);
  cp16_g2l(swr + 8, &Bs[0][lrow][lcol + 8]);
  async_fence();
  __syncthreads();

  for (int k0 = 0; k0 < D_; k0 += BK) {
    int buf = (k0 / BK) & 1;
    if (k0 + BK < D_) {                    // prefetch next tile into other buffer
      cp16_g2l(sxr + (k0 + BK),     &As[buf ^ 1][lrow][lcol]);
      cp16_g2l(sxr + (k0 + BK) + 8, &As[buf ^ 1][lrow][lcol + 8]);
      cp16_g2l(swr + (k0 + BK),     &Bs[buf ^ 1][lrow][lcol]);
      cp16_g2l(swr + (k0 + BK) + 8, &Bs[buf ^ 1][lrow][lcol + 8]);
    }

    v16bf af[2], bfr[4];
#pragma unroll
    for (int t = 0; t < 2; ++t)
      af[t] = load_a_frag(&As[buf][wm * 32 + t * 16][0], BK, lane);
#pragma unroll
    for (int u = 0; u < 4; ++u)
      bfr[u] = load_b_frag(&Bs[buf][wn * 64 + u * 16][0], BK, lane);
#pragma unroll
    for (int t = 0; t < 2; ++t)
#pragma unroll
      for (int u = 0; u < 4; ++u)
        acc[t][u] = wmma_bf16(af[t], bfr[u], acc[t][u]);

    async_fence();
    __syncthreads();
  }

  // epilogue: + bias, (Q only) * 1/8, store bf16
  int nlo = lane & 15, h8 = (lane >> 4) << 3;
  float sc = (proj == 0) ? QSCALE : 1.0f;
#pragma unroll
  for (int u = 0; u < 4; ++u) {
    int n = n0 + wn * 64 + u * 16 + nlo;
    float bvs = bias[n];
#pragma unroll
    for (int t = 0; t < 2; ++t) {
#pragma unroll
      for (int r = 0; r < 8; ++r) {
        int m = m0 + wm * 32 + t * 16 + r + h8;
        Y[(size_t)m * D_ + n] = f2bf((acc[t][u][r] + bvs) * sc);
      }
    }
  }
}

// ---------------------------------------------------------------------------
// Kernel 4: flash attention.  grid (S/64, B*H), 128 threads = 4 waves.
// Wave handles 16 query rows. Computes S^T = K*Q^T so that the C-layout of
// S^T maps register-only onto the A-fragment layout of P for O += P*V.
// K tile staged via async global->LDS; V staged transposed (scalar stores).
// ---------------------------------------------------------------------------
__global__ __launch_bounds__(128)
void attn(const unsigned short* __restrict__ Q, const unsigned short* __restrict__ K,
          const unsigned short* __restrict__ V, const int* __restrict__ mask,
          float* __restrict__ out) {
  __shared__ unsigned short Ks[32][HW_];   // keys x w
  __shared__ unsigned short Vt[HW_][32];   // w x keys (transposed)
  __shared__ float smadd[32];              // mask addend per key

  int bh = blockIdx.y, b = bh / H_, h = bh % H_;
  int q0 = blockIdx.x * 64;
  int tid = threadIdx.x, wave = tid >> 5, lane = tid & 31;
  int qrow0 = q0 + wave * 16;
  int nlo = lane & 15, h8 = (lane >> 4) << 3;

  const unsigned short* Qbase = Q + (size_t)(b * S_) * D_ + h * HW_;
  const unsigned short* Kbase = K + (size_t)(b * S_) * D_ + h * HW_;
  const unsigned short* Vbase = V + (size_t)(b * S_) * D_ + h * HW_;
  const int* mrow = mask + b * S_;

  // Q fragments (B-matrix of S^T GEMM): lane n = qrow, w 0-7,16-23 (lanes 0-15) / 8-15,24-31 (lanes 16-31).
  v16bf qb[2];
#pragma unroll
  for (int f = 0; f < 2; ++f)
    qb[f] = load_b_frag(Qbase + (size_t)qrow0 * D_ + f * 32, D_, lane);

  v8f o[4];
#pragma unroll
  for (int wt = 0; wt < 4; ++wt) o[wt] = (v8f){0,0,0,0,0,0,0,0};
  float m_i = -1e30f, l_i = 0.0f;   // live at lane = qrow (duplicated lane^16)

  int lkey = tid >> 2, lseg = (tid & 3) * 16;
  for (int kv0 = 0; kv0 < S_; kv0 += 32) {
    __syncthreads();
    {
      const unsigned short* pk = Kbase + (size_t)(kv0 + lkey) * D_ + lseg;
      cp16_g2l(pk,     &Ks[lkey][lseg]);       // async -> ASYNCcnt
      cp16_g2l(pk + 8, &Ks[lkey][lseg + 8]);
      const unsigned short* pv = Vbase + (size_t)(kv0 + lkey) * D_ + lseg;
#pragma unroll
      for (int j = 0; j < 16; ++j) Vt[lseg + j][lkey] = pv[j];   // transpose
    }
    if (tid < 32) smadd[tid] = mrow[kv0 + tid] ? 0.0f : -10000.0f;
    async_fence();
    __syncthreads();

    if (kv0 + 32 < S_) {   // prefetch next block -> global_prefetch_b8
      __builtin_prefetch(Kbase + (size_t)(kv0 + 32 + lkey) * D_ + lseg, 0, 1);
      __builtin_prefetch(Vbase + (size_t)(kv0 + 32 + lkey) * D_ + lseg, 0, 1);
    }

    // S^T tiles: M = keys (2 tiles of 16), N = qrows, K-dim = w (64 = 2 steps)
    v8f st[2];
#pragma unroll
    for (int kt = 0; kt < 2; ++kt) {
      v8f c = (v8f){0,0,0,0,0,0,0,0};
      c = wmma_bf16(load_a_frag(&Ks[kt * 16][0],  HW_, lane), qb[0], c);
      c = wmma_bf16(load_a_frag(&Ks[kt * 16][32], HW_, lane), qb[1], c);
      st[kt] = c;
    }

    // mask + block row-max (row = qrow = lane&15; keys spread over vgprs/lane-half)
    float bmax = -1e30f;
#pragma unroll
    for (int kt = 0; kt < 2; ++kt)
#pragma unroll
      for (int r = 0; r < 8; ++r) {
        st[kt][r] += smadd[kt * 16 + h8 + r];
        bmax = fmaxf(bmax, st[kt][r]);
      }
    bmax = fmaxf(bmax, __shfl_xor(bmax, 16, 32));
    float mnew  = fmaxf(m_i, bmax);
    float alpha = __expf(m_i - mnew);

    // P = exp(S - mnew); register-only remap of S^T C-layout -> P A-fragment
    float psum = 0.0f;
    v16u16 pu;
#pragma unroll
    for (int kt = 0; kt < 2; ++kt)
#pragma unroll
      for (int r = 0; r < 8; ++r) {
        float p = __expf(st[kt][r] - mnew);
        psum += p;
        pu[kt * 8 + r] = f2bf(p);
      }
    psum += __shfl_xor(psum, 16, 32);
    l_i = l_i * alpha + psum;
    m_i = mnew;

    // rescale O: row qrow = r + h8 lives at lane (r + h8)
#pragma unroll
    for (int r = 0; r < 8; ++r) {
      float ar = __shfl(alpha, h8 + r, 32);
#pragma unroll
      for (int wt = 0; wt < 4; ++wt) o[wt][r] *= ar;
    }

    v16bf pa = __builtin_bit_cast(v16bf, pu);
#pragma unroll
    for (int wt = 0; wt < 4; ++wt)
      o[wt] = wmma_bf16(pa, load_b_frag(&Vt[wt * 16][0], 32, lane), o[wt]);
  }

  // normalize + store (f32 output)
  float linv = 1.0f / l_i;
#pragma unroll
  for (int r = 0; r < 8; ++r) {
    float lr = __shfl(linv, h8 + r, 32);
    int m = qrow0 + h8 + r;
#pragma unroll
    for (int wt = 0; wt < 4; ++wt)
      out[(size_t)(b * S_ + m) * D_ + h * HW_ + wt * 16 + nlo] = o[wt][r] * lr;
  }
}

// ---------------------------------------------------------------------------
// Launch
// ---------------------------------------------------------------------------
extern "C" void kernel_launch(void* const* d_in, const int* in_sizes, int n_in,
                              void* d_out, int out_size, void* d_ws, size_t ws_size,
                              hipStream_t stream) {
  (void)in_sizes; (void)n_in; (void)out_size; (void)ws_size;
  const float* x  = (const float*)d_in[0];
  const int* mask = (const int*)d_in[1];
  const float* Wq = (const float*)d_in[2];
  const float* bq = (const float*)d_in[3];
  const float* Aq = (const float*)d_in[4];
  const float* Bq = (const float*)d_in[5];
  const float* Wk = (const float*)d_in[6];
  const float* bk = (const float*)d_in[7];
  const float* Wv = (const float*)d_in[8];
  const float* bv = (const float*)d_in[9];
  const float* Av = (const float*)d_in[10];
  const float* Bv = (const float*)d_in[11];

  const size_t M = (size_t)B_ * S_;           // 4096
  unsigned short* xb = (unsigned short*)d_ws;             // M*D
  unsigned short* Wb = xb + M * D_;                       // 3*D*D
  unsigned short* Y  = Wb + (size_t)3 * D_ * D_;          // 3*M*D (Q,K,V)

  prep_weights<<<dim3((D_ * D_) / 256, 3), 256, 0, stream>>>(Wq, Aq, Bq, Wk, Wv,
                                                             Av, Bv, Wb);
  convert_x<<<(unsigned)((M * D_) / 256), 256, 0, stream>>>(x, xb);
  proj_gemm<<<dim3(D_ / BN, M / BM, 3), 256, 0, stream>>>(xb, Wb, bq, bk, bv, Y);
  attn<<<dim3(S_ / 64, B_ * H_), 128, 0, stream>>>(Y, Y + M * D_, Y + 2 * M * D_,
                                                   mask, (float*)d_out);
}
